// ParallelRetention_25709674234637
// MI455X (gfx1250) — matrix operand order repaired and checked by reading, not compiled, not run on hardware
//
#include <hip/hip_runtime.h>
#include <hip/hip_bf16.h>
#include <cstdint>

// Problem constants (B,T,D,H from reference)
constexpr int Bb = 2;
constexpr int Tt = 2048;
constexpr int Dd = 512;
constexpr int Hh = 8;
constexpr int HD = 64;          // head dim
constexpr int Mm = Bb * Tt;     // 4096 rows for projections
constexpr float EPS = 1e-5f;

typedef __attribute__((ext_vector_type(16))) __bf16          v16bf;
typedef __attribute__((ext_vector_type(8)))  float           v8f;
typedef __attribute__((ext_vector_type(8)))  unsigned short  ush8;

// ---------- helpers ----------
static __device__ __forceinline__ unsigned short f2bf(float f) {
    unsigned int u = __float_as_uint(f);
    unsigned int r = u + 0x7FFFu + ((u >> 16) & 1u);   // round-to-nearest-even
    return (unsigned short)(r >> 16);
}

// Build a 16-element bf16 operand vector from two 16-byte chunks.
static __device__ __forceinline__ v16bf ld2x8(const unsigned short* p0,
                                              const unsigned short* p1) {
    union { ush8 h[2]; v16bf v; } u;
    u.h[0] = *(const ush8*)p0;
    u.h[1] = *(const ush8*)p1;
    return u.v;
}

static __device__ __forceinline__ v8f wmma_bf16(v16bf a, v16bf b, v8f c) {
    // (neg_a, A, neg_b, B, c_mod, C, reuse_a, reuse_b)
    return __builtin_amdgcn_wmma_f32_16x16x32_bf16(false, a, false, b,
                                                   (short)0, c, false, false);
}

// ---------- kernel 1: f32 -> bf16 convert ----------
__global__ void pr_cvt_bf16(const float* __restrict__ src,
                            unsigned short* __restrict__ dst, int n) {
    int i = blockIdx.x * blockDim.x + threadIdx.x;
    for (; i < n; i += gridDim.x * blockDim.x) dst[i] = f2bf(src[i]);
}

// ---------- kernel 2: QKV projection (Y = x @ W^T + b), WMMA bf16 ----------
// grid (Mm/16, Hh, 3), block 32: one wave computes a 16x64 tile = one head slice.
__global__ __launch_bounds__(32)
void pr_proj(const unsigned short* __restrict__ xh,   // [Mm, Dd] bf16
             const unsigned short* __restrict__ wh,   // [3][Dd, Dd] bf16 (row = out ch)
             const float* __restrict__ bq,
             const float* __restrict__ bk,
             const float* __restrict__ bv,
             unsigned short* __restrict__ qh,          // [B,H,T,HD] bf16
             unsigned short* __restrict__ kh,          // [B,H,T,HD] bf16
             unsigned short* __restrict__ vt)          // [B,H,HD,T] bf16 (V transposed)
{
    const int lane = threadIdx.x;
    const int col  = lane & 15;
    const int hi   = lane >> 4;
    const int m0   = blockIdx.x * 16;
    const int h    = blockIdx.y;          // head (o0 = h*64)
    const int mat  = blockIdx.z;
    const int o0   = h * HD;

    const unsigned short* wm  = wh + (size_t)mat * Dd * Dd;
    const float* bias = (mat == 0) ? bq : (mat == 1) ? bk : bv;

    // A operand: row m0+col, 16 contiguous K at hi*16 per 32-chunk
    const unsigned short* ap  = xh + (size_t)(m0 + col) * Dd + hi * 16;
    // B operands: W rows = out channels o0 + j*16 + col, 16 contiguous K at hi*16
    const unsigned short* bp0 = wm + (size_t)(o0 +  0 + col) * Dd + hi * 16;
    const unsigned short* bp1 = wm + (size_t)(o0 + 16 + col) * Dd + hi * 16;
    const unsigned short* bp2 = wm + (size_t)(o0 + 32 + col) * Dd + hi * 16;
    const unsigned short* bp3 = wm + (size_t)(o0 + 48 + col) * Dd + hi * 16;

    v8f a0 = {}, a1 = {}, a2 = {}, a3 = {};
    #pragma unroll 2
    for (int dk = 0; dk < Dd; dk += 32) {
        v16bf A = ld2x8(ap + dk, ap + dk + 8);           // shared across 4 tiles
        a0 = wmma_bf16(A, ld2x8(bp0 + dk, bp0 + dk + 8), a0);
        a1 = wmma_bf16(A, ld2x8(bp1 + dk, bp1 + dk + 8), a1);
        a2 = wmma_bf16(A, ld2x8(bp2 + dk, bp2 + dk + 8), a2);
        a3 = wmma_bf16(A, ld2x8(bp3 + dk, bp3 + dk + 8), a3);
    }

    const float bv0 = bias[o0 +  0 + col];
    const float bv1 = bias[o0 + 16 + col];
    const float bv2 = bias[o0 + 32 + col];
    const float bv3 = bias[o0 + 48 + col];

    #pragma unroll
    for (int r = 0; r < 8; ++r) {
        const int m  = m0 + r + hi * 8;        // row in [Mm]
        const int bb = m >> 11;                // batch (T=2048)
        const int t  = m & (Tt - 1);
        const size_t g = (size_t)(bb * Hh + h);
        const unsigned short u0 = f2bf(a0[r] + bv0);
        const unsigned short u1 = f2bf(a1[r] + bv1);
        const unsigned short u2 = f2bf(a2[r] + bv2);
        const unsigned short u3 = f2bf(a3[r] + bv3);
        if (mat == 0) {
            unsigned short* p = qh + (g * Tt + t) * HD + col;
            p[0] = u0; p[16] = u1; p[32] = u2; p[48] = u3;
        } else if (mat == 1) {
            unsigned short* p = kh + (g * Tt + t) * HD + col;
            p[0] = u0; p[16] = u1; p[32] = u2; p[48] = u3;
        } else {
            unsigned short* p = vt + (g * HD + col) * Tt + t;  // transposed
            p[0 * 16 * Tt] = u0; p[1 * 16 * Tt] = u1;
            p[2 * 16 * Tt] = u2; p[3 * 16 * Tt] = u3;
        }
    }
}

// ---------- kernel 3: fused retention  O = (QK^T * decay) V ----------
// grid (Tt/16, Hh, Bb), block 32 (one wave per 16-row q tile)
__global__ __launch_bounds__(32)
void pr_retention(const unsigned short* __restrict__ qh,
                  const unsigned short* __restrict__ kh,
                  const unsigned short* __restrict__ vt,
                  const float* __restrict__ gamma_p,
                  float* __restrict__ Obuf)             // [B*H, T, HD] f32 pre-norm
{
    __shared__ alignas(16) unsigned short Plds[16 * 40];  // 16x32 bf16, row stride 40

    const int lane = threadIdx.x;
    const int col  = lane & 15;
    const int hi   = lane >> 4;
    const int g    = blockIdx.z * Hh + blockIdx.y;        // (b,h) group
    const int q0   = blockIdx.x * 16;

    const unsigned short* qb = qh + (size_t)g * Tt * HD;
    const unsigned short* kb = kh + (size_t)g * Tt * HD;
    const unsigned short* vb = vt + (size_t)g * HD * Tt;

    const float lg = log2f(gamma_p[0]);

    // Row decay factors gamma^(q-q0), computed once (keeps trans pipe free for WMMA)
    float rowf[8];
    #pragma unroll
    for (int r = 0; r < 8; ++r)
        rowf[r] = exp2f((float)(r + hi * 8) * lg);

    // Q tile 16x64 held as two A operands (K = 0..31 and 32..63)
    const unsigned short* qp = qb + (size_t)(q0 + col) * HD + hi * 16;
    const v16bf qa0 = ld2x8(qp,      qp + 8);
    const v16bf qa1 = ld2x8(qp + 32, qp + 40);

    v8f o0 = {}, o1 = {}, o2 = {}, o3 = {};
    const int qend = q0 + 15;

    for (int k0 = 0; k0 <= qend; k0 += 32) {
        if (k0 + 32 <= qend) {   // prefetch next chunk (global_prefetch_b8)
            __builtin_prefetch(kb + (size_t)(k0 + 32 + col) * HD, 0, 0);
            __builtin_prefetch(vb + (size_t)col * Tt + (k0 + 32), 0, 0);
        }

        // ---- S = Q * K^T : two 16x16 sub-tiles (k-cols 0..15 and 16..31) ----
        v8f s0 = {}, s1 = {};
        {
            const unsigned short* kp = kb + (size_t)(k0 + col) * HD + hi * 16;
            s0 = wmma_bf16(qa0, ld2x8(kp,      kp + 8),  s0);
            s0 = wmma_bf16(qa1, ld2x8(kp + 32, kp + 40), s0);
            const unsigned short* kp1 = kb + (size_t)(k0 + 16 + col) * HD + hi * 16;
            s1 = wmma_bf16(qa0, ld2x8(kp1,      kp1 + 8),  s1);
            s1 = wmma_bf16(qa1, ld2x8(kp1 + 32, kp1 + 40), s1);
        }

        // ---- factored causal decay: gamma^(q-k) = rowf[q-q0] * gamma^(q0-k) ----
        const int kg0 = k0 + col;
        const int kg1 = k0 + 16 + col;
        const float colf0 = exp2f((float)(q0 - kg0) * lg);   // only 2 exp2 per chunk
        const float colf1 = exp2f((float)(q0 - kg1) * lg);
        #pragma unroll
        for (int r = 0; r < 8; ++r) {
            const int qg = q0 + r + hi * 8;        // C layout: row = r + 8*hi
            float p0 = (qg >= kg0) ? s0[r] * (rowf[r] * colf0) : 0.0f;
            float p1 = (qg >= kg1) ? s1[r] * (rowf[r] * colf1) : 0.0f;
            Plds[(r + hi * 8) * 40 + col]      = f2bf(p0);
            Plds[(r + hi * 8) * 40 + 16 + col] = f2bf(p1);
        }
        __syncthreads();

        // re-load P in A-operand layout (K = 32)
        v16bf pa;
        {
            const unsigned short* pp = &Plds[col * 40 + hi * 16];
            pa = ld2x8(pp, pp + 8);
        }

        // ---- O += P * V : four 16x16 hd tiles; V stored transposed ----
        {
            const unsigned short* vp0 = vb + (size_t)(0 * 16 + col) * Tt + k0 + hi * 16;
            const unsigned short* vp1 = vb + (size_t)(1 * 16 + col) * Tt + k0 + hi * 16;
            const unsigned short* vp2 = vb + (size_t)(2 * 16 + col) * Tt + k0 + hi * 16;
            const unsigned short* vp3 = vb + (size_t)(3 * 16 + col) * Tt + k0 + hi * 16;
            o0 = wmma_bf16(pa, ld2x8(vp0, vp0 + 8), o0);
            o1 = wmma_bf16(pa, ld2x8(vp1, vp1 + 8), o1);
            o2 = wmma_bf16(pa, ld2x8(vp2, vp2 + 8), o2);
            o3 = wmma_bf16(pa, ld2x8(vp3, vp3 + 8), o3);
        }
        __syncthreads();   // protect Plds before next chunk overwrites it
    }

    // ---- store pre-norm O (f32) ----
    float* ob = Obuf + (size_t)g * Tt * HD;
    #pragma unroll
    for (int r = 0; r < 8; ++r) {
        const int qg = q0 + r + hi * 8;
        ob[(size_t)qg * HD +  0 + col] = o0[r];
        ob[(size_t)qg * HD + 16 + col] = o1[r];
        ob[(size_t)qg * HD + 32 + col] = o2[r];
        ob[(size_t)qg * HD + 48 + col] = o3[r];
    }
}

// ---------- kernel 4: GroupNorm statistics per (b,h) ----------
__global__ __launch_bounds__(256)
void pr_gnstats(const float* __restrict__ Obuf, float* __restrict__ stats) {
    const int g = blockIdx.x;                    // 0..15
    const int N = Tt * HD;                       // 131072
    const float* p = Obuf + (size_t)g * N;
    float s = 0.f, s2 = 0.f;
    for (int i = threadIdx.x; i < N; i += 256) {
        float v = p[i];
        s += v; s2 += v * v;
    }
    __shared__ float sh[256], sh2[256];
    sh[threadIdx.x] = s; sh2[threadIdx.x] = s2;
    __syncthreads();
    for (int off = 128; off > 0; off >>= 1) {
        if (threadIdx.x < off) {
            sh[threadIdx.x]  += sh[threadIdx.x + off];
            sh2[threadIdx.x] += sh2[threadIdx.x + off];
        }
        __syncthreads();
    }
    if (threadIdx.x == 0) {
        float mean = sh[0] / (float)N;
        float var  = fmaxf(sh2[0] / (float)N - mean * mean, 0.0f);
        stats[2 * g]     = mean;
        stats[2 * g + 1] = rsqrtf(var + EPS);
    }
}

// ---------- kernel 5: normalize + affine, write [B,T,D] f32 ----------
__global__ __launch_bounds__(256)
void pr_gnapply(const float* __restrict__ Obuf, const float* __restrict__ stats,
                const float* __restrict__ gw, const float* __restrict__ gb,
                float* __restrict__ out) {
    const int n = Bb * Tt * Dd;
    int idx = blockIdx.x * 256 + threadIdx.x;
    if (idx >= n) return;
    const int c  = idx & (Dd - 1);
    const int t  = (idx / Dd) & (Tt - 1);
    const int bb = idx / (Dd * Tt);
    const int h  = c >> 6;
    const int hd = c & (HD - 1);
    const int g  = bb * Hh + h;
    const float mean = stats[2 * g];
    const float rstd = stats[2 * g + 1];
    const float v = Obuf[((size_t)g * Tt + t) * HD + hd];
    out[idx] = (v - mean) * rstd * gw[c] + gb[c];
}

// ---------- launch ----------
extern "C" void kernel_launch(void* const* d_in, const int* in_sizes, int n_in,
                              void* d_out, int out_size, void* d_ws, size_t ws_size,
                              hipStream_t stream) {
    (void)in_sizes; (void)n_in; (void)out_size; (void)ws_size;
    const float* x   = (const float*)d_in[0];
    const float* Wq  = (const float*)d_in[1];
    const float* bq  = (const float*)d_in[2];
    const float* Wk  = (const float*)d_in[3];
    const float* bk  = (const float*)d_in[4];
    const float* Wv  = (const float*)d_in[5];
    const float* bv  = (const float*)d_in[6];
    const float* gnw = (const float*)d_in[7];
    const float* gnb = (const float*)d_in[8];
    const float* gam = (const float*)d_in[9];

    char* ws = (char*)d_ws;
    const size_t nX  = (size_t)Mm * Dd;          // 2,097,152
    const size_t nW  = (size_t)Dd * Dd;          // 262,144
    size_t off = 0;
    unsigned short* xh = (unsigned short*)(ws + off); off += nX * 2;        // 4 MB
    unsigned short* wh = (unsigned short*)(ws + off); off += 3 * nW * 2;    // 1.5 MB
    unsigned short* qh = (unsigned short*)(ws + off); off += nX * 2;        // 4 MB
    unsigned short* kh = (unsigned short*)(ws + off); off += nX * 2;        // 4 MB
    unsigned short* vt = (unsigned short*)(ws + off); off += nX * 2;        // 4 MB
    float*          Ob = (float*)(ws + off);          off += nX * 4;        // 8 MB
    float*          st = (float*)(ws + off);                                 // 128 B

    // 1) convert inputs to bf16
    pr_cvt_bf16<<<(int)((nX + 255) / 256), 256, 0, stream>>>(x,  xh, (int)nX);
    pr_cvt_bf16<<<(int)((nW + 255) / 256), 256, 0, stream>>>(Wq, wh,          (int)nW);
    pr_cvt_bf16<<<(int)((nW + 255) / 256), 256, 0, stream>>>(Wk, wh + nW,     (int)nW);
    pr_cvt_bf16<<<(int)((nW + 255) / 256), 256, 0, stream>>>(Wv, wh + 2 * nW, (int)nW);

    // 2) QKV projections (WMMA, 16x64 tile per wave = one head slice)
    pr_proj<<<dim3(Mm / 16, Hh, 3), 32, 0, stream>>>(xh, wh, bq, bk, bv,
                                                     qh, kh, vt);
    // 3) fused retention (WMMA)
    pr_retention<<<dim3(Tt / 16, Hh, Bb), 32, 0, stream>>>(qh, kh, vt, gam, Ob);

    // 4) GroupNorm
    pr_gnstats<<<Bb * Hh, 256, 0, stream>>>(Ob, st);
    pr_gnapply<<<(Bb * Tt * Dd + 255) / 256, 256, 0, stream>>>(Ob, st, gnw, gnb,
                                                               (float*)d_out);
}
